// S4Encoder_15968688407047
// MI455X (gfx1250) — matrix operand order repaired and checked by reading, not compiled, not run on hardware
//
#include <hip/hip_runtime.h>
#include <hip/hip_bf16.h>

typedef __attribute__((ext_vector_type(16))) _Float16 v16h;
typedef __attribute__((ext_vector_type(8)))  _Float16 v8h;
typedef __attribute__((ext_vector_type(8)))  float    v8f;
typedef __attribute__((ext_vector_type(4)))  unsigned int u32x4;
typedef __attribute__((ext_vector_type(4)))  int          i32x4;
typedef __attribute__((ext_vector_type(8)))  int          i32x8;

#define B_     16
#define L_     5000
#define DIN_   12
#define H_     256
#define NM_    32      // complex modes per channel
#define NL_    4
#define Q_     64      // chunk length
#define CCH_   79      // ceil(L/Q)
#define BC_    (B_*CCH_)   // 1264 chunk-columns
#define NTSL_  8       // nt slices for gemmA/B occupancy
#define NTPER_ 10      // ceil(79/8)

// ---------------------------------------------------------------------------
// TDM: flat 1-D copy global -> LDS (data_size = 8B units). Issue from ONE wave.
// ---------------------------------------------------------------------------
__device__ __forceinline__ void tdm_load_1d(unsigned lds_off, const void* gptr,
                                            unsigned nbytes) {
  unsigned long long ga = (unsigned long long)(uintptr_t)gptr;
  unsigned elems = nbytes >> 3;            // 8-byte elements
  u32x4 g0;
  g0[0] = 1u;                              // count=1, user-mode descriptor
  g0[1] = lds_off;                         // lds_addr (bytes)
  g0[2] = (unsigned)(ga & 0xFFFFFFFFu);    // global_addr[31:0]
  g0[3] = (unsigned)((ga >> 32) & 0x1FFFFFFu) | (2u << 30);  // addr[56:32] | type=2
  i32x8 g1;
  g1[0] = (int)(3u << 16);                                   // data_size=8B
  g1[1] = (int)((elems & 0xFFFFu) << 16);                    // tensor_dim0[15:0]
  g1[2] = (int)(((elems >> 16) & 0xFFFFu) | (1u << 16));     // tensor_dim0[31:16] | tensor_dim1=1
  g1[3] = (int)((elems & 0xFFFFu) << 16);                    // tile_dim0
  g1[4] = 0;                                                 // tile_dim1=0, tile_dim2=0
  g1[5] = (int)elems;                                        // tensor_dim0_stride lo
  g1[6] = 0;
  g1[7] = 0;
  i32x4 gz = {0, 0, 0, 0};
#if defined(__clang_major__) && (__clang_major__ >= 23)
  i32x8 gz8 = {0, 0, 0, 0, 0, 0, 0, 0};
  __builtin_amdgcn_tensor_load_to_lds(g0, g1, gz, gz, gz8, 0);
#else
  __builtin_amdgcn_tensor_load_to_lds(g0, g1, gz, gz, 0);
#endif
}

// ---------------------------------------------------------------------------
// Input projection: x(B,L,12) @ W_in(12,H) + b_in -> xres(B,L,H)
// ---------------------------------------------------------------------------
__global__ void k_inproj(const float* __restrict__ x, const float* __restrict__ W,
                         const float* __restrict__ b, float* __restrict__ out) {
  int idx = blockIdx.x * blockDim.x + threadIdx.x;
  if (idx >= B_ * L_ * H_) return;
  int h = idx % H_;
  int row = idx / H_;
  const float* xr = x + (long)row * DIN_;
  float acc = b[h];
#pragma unroll
  for (int k = 0; k < DIN_; ++k) acc += xr[k] * W[k * H_ + h];
  out[idx] = acc;
}

// ---------------------------------------------------------------------------
// LayerNorm over H=256, one block per row; safe in-place.
// ---------------------------------------------------------------------------
__global__ void k_layernorm(const float* __restrict__ x, const float* __restrict__ g,
                            const float* __restrict__ bt, float* __restrict__ out) {
  int row = blockIdx.x;
  int t = threadIdx.x;
  float v = x[(long)row * H_ + t];
  __shared__ float sm[H_];
  sm[t] = v;
  __syncthreads();
  for (int s = 128; s > 0; s >>= 1) { if (t < s) sm[t] += sm[t + s]; __syncthreads(); }
  float mean = sm[0] * (1.0f / H_);
  __syncthreads();
  float d = v - mean;
  sm[t] = d * d;
  __syncthreads();
  for (int s = 128; s > 0; s >>= 1) { if (t < s) sm[t] += sm[t + s]; __syncthreads(); }
  float var = sm[0] * (1.0f / H_);
  out[(long)row * H_ + t] = d * rsqrtf(var + 1e-5f) * g[t] + bt[t];
}

// ---------------------------------------------------------------------------
// Pack u: z(B,L,H) f32 -> uT[h][col=(b,c)][j] f16 (chunk-column major, zero pad)
// ---------------------------------------------------------------------------
__global__ void k_pack_u(const float* __restrict__ z, _Float16* __restrict__ uT) {
  long idx = (long)blockIdx.x * blockDim.x + threadIdx.x;
  if (idx >= (long)H_ * BC_ * Q_) return;
  int j = (int)(idx % Q_);
  long r = idx / Q_;
  int col = (int)(r % BC_);
  int h = (int)(r / BC_);
  int b = col / CCH_, c = col % CCH_;
  int l = c * Q_ + j;
  float v = (l < L_) ? z[((long)(b * L_ + l)) * H_ + h] : 0.0f;
  uT[idx] = (_Float16)v;
}

// ---------------------------------------------------------------------------
// SSM precompute per channel h (one wave, lane = mode n)
// ---------------------------------------------------------------------------
__global__ void k_ssm_pre(const float* __restrict__ log_dt, const float* __restrict__ logA_re,
                          const float* __restrict__ A_im, const float* __restrict__ Cp,
                          _Float16* __restrict__ Amat, _Float16* __restrict__ Wout,
                          float* __restrict__ lamQ) {
  int h = blockIdx.x;
  int n = threadIdx.x;  // 0..31
  float dt  = __expf(log_dt[h]);
  float Are = -__expf(logA_re[h * NM_ + n]);
  float Aim = A_im[h * NM_ + n];
  float Cre = Cp[((long)h * NM_ + n) * 2 + 0];
  float Cim = Cp[((long)h * NM_ + n) * 2 + 1];
  float dre = Are * dt, dim = Aim * dt;
  float er  = __expf(dre);
  float lre = er * __cosf(dim), lim = er * __sinf(dim);   // lam = exp(dt*A)
  float numr = lre - 1.0f, numi = lim;
  float den = Are * Are + Aim * Aim;
  float qr = (numr * Are + numi * Aim) / den;
  float qi = (numi * Are - numr * Aim) / den;
  float Ctr = Cre * qr - Cim * qi;
  float Cti = Cre * qi + Cim * qr;

  __shared__ float Ktaps[Q_];
  _Float16* Ah = Amat + (long)h * 128 * Q_;
  _Float16* Wh = Wout + (long)h * Q_ * 64;

  float pr = Ctr, pi = Cti;   // Ct * lam^l
  float wr = 1.0f, wi = 0.0f; // lam^l
  for (int l = 0; l < Q_; ++l) {
    float val = 2.0f * pr;
    for (int off = 16; off > 0; off >>= 1) val += __shfl_down(val, off, 32);
    if (n == 0) Ktaps[l] = val;
    float npr = pr * lre - pi * lim;
    float npi = pr * lim + pi * lre;
    pr = npr; pi = npi;
    Wh[l * 64 + n]       = (_Float16)( 2.0f * pr);
    Wh[l * 64 + NM_ + n] = (_Float16)(-2.0f * pi);
    Ah[(64 + n)       * Q_ + (Q_ - 1 - l)] = (_Float16)wr;
    Ah[(64 + NM_ + n) * Q_ + (Q_ - 1 - l)] = (_Float16)wi;
    float nwr = wr * lre - wi * lim;
    float nwi = wr * lim + wi * lre;
    wr = nwr; wi = nwi;
  }
  lamQ[((long)h * NM_ + n) * 2 + 0] = wr;  // lam^Q
  lamQ[((long)h * NM_ + n) * 2 + 1] = wi;
  __syncthreads();
  for (int e = n; e < Q_ * Q_; e += 32) {
    int i = e / Q_, j = e % Q_;
    Ah[i * Q_ + j] = (_Float16)((i >= j) ? Ktaps[i - j] : 0.0f);
  }
}

// Build a 16x32 f16 A-fragment from two contiguous 8-half runs (ISA 7.12.2)
__device__ __forceinline__ v16h frag_from_runs(v8h r0, v8h r1) {
  v16h a;
#pragma unroll
  for (int i = 0; i < 8; ++i) { a[i] = r0[i]; a[8 + i] = r1[i]; }
  return a;
}

// ---------------------------------------------------------------------------
// gemmA: per h, [Toeplitz ; Win_state] (128x64) @ uT_h (64x1264).
// Workgroup = 8 waves (one M-tile each); Amat[h] staged via TDM into LDS,
// A-fragments preloaded in registers and reused across all nt in this slice.
// grid = (H_, NTSL_)
// ---------------------------------------------------------------------------
__global__ __launch_bounds__(256) void k_gemmA(const _Float16* __restrict__ Amat,
                                               const _Float16* __restrict__ uT,
                                               float* __restrict__ yconv,
                                               float* __restrict__ scon) {
  __shared__ __align__(128) _Float16 smA[128 * Q_];   // 16 KB
  int h = blockIdx.x;
  if (threadIdx.x < 32) {
    tdm_load_1d((unsigned)(uintptr_t)&smA[0], Amat + (size_t)h * 128 * Q_, 128 * Q_ * 2);
    __builtin_amdgcn_s_wait_tensorcnt(0);
  }
  __syncthreads();
  int wave = threadIdx.x >> 5;   // M-tile 0..7
  int lane = threadIdx.x & 31;
  int mi = lane & 15;
  int hi = lane >> 4;
  int koff = hi ? 16 : 0;
  int klo  = hi ? 16 : 0;
  const _Float16* pa = &smA[(wave * 16 + mi) * Q_ + koff];
  v16h a0 = frag_from_runs(*(const v8h*)(pa),      *(const v8h*)(pa + 8));
  v16h a1 = frag_from_runs(*(const v8h*)(pa + 32), *(const v8h*)(pa + 40));

  int nt0 = blockIdx.y * NTPER_;
  int nt1 = nt0 + NTPER_; if (nt1 > CCH_) nt1 = CCH_;
  for (int nt = nt0; nt < nt1; ++nt) {
    const _Float16* Uh = uT + ((size_t)h * BC_ + (size_t)nt * 16) * Q_;
    v16h b0 = *(const v16h*)(Uh + (long)mi * Q_ + 0  + klo);
    v16h b1 = *(const v16h*)(Uh + (long)mi * Q_ + 32 + klo);
    v8f c = {};
    c = __builtin_amdgcn_wmma_f32_16x16x32_f16(false, a0, false, b0, (short)0, c, false, false);
    c = __builtin_amdgcn_wmma_f32_16x16x32_f16(false, a1, false, b1, (short)0, c, false, false);
    int rowbase = wave * 16 + (hi ? 8 : 0);
    int ncol = nt * 16 + mi;
    int b_ = ncol / CCH_, cch = ncol % CCH_;
    if (wave < 4) {           // intra-chunk output rows 0..63
#pragma unroll
      for (int r = 0; r < 8; ++r) {
        int l = cch * Q_ + rowbase + r;
        if (l < L_) yconv[((long)(b_ * L_ + l)) * H_ + h] = c[r];
      }
    } else {                  // state-contribution rows 64..127
      long base = ((long)h * BC_ + ncol) * 64 + (rowbase - 64);
#pragma unroll
      for (int r = 0; r < 8; ++r) scon[base + r] = c[r];
    }
  }
}

// ---------------------------------------------------------------------------
// Chunk-boundary state scan: S[c] = lam^Q * S[c-1] + scon[c]; emit S[c-1] f16
// ---------------------------------------------------------------------------
__global__ void k_scan(const float* __restrict__ scon, const float* __restrict__ lamQ,
                       _Float16* __restrict__ sprevT) {
  int idx = blockIdx.x * blockDim.x + threadIdx.x;
  if (idx >= B_ * H_ * NM_) return;
  int n = idx % NM_;
  int r = idx / NM_;
  int h = r % H_;
  int b = r / H_;
  float lr = lamQ[((long)h * NM_ + n) * 2 + 0];
  float li = lamQ[((long)h * NM_ + n) * 2 + 1];
  float sr = 0.0f, si = 0.0f;
  for (int c = 0; c < CCH_; ++c) {
    long base = ((long)h * BC_ + (long)b * CCH_ + c) * 64;
    sprevT[base + n]       = (_Float16)sr;
    sprevT[base + NM_ + n] = (_Float16)si;
    float ar = scon[base + n], ai = scon[base + NM_ + n];
    float nsr = lr * sr - li * si + ar;
    float nsi = lr * si + li * sr + ai;
    sr = nsr; si = nsi;
  }
}

// ---------------------------------------------------------------------------
// gemmB: per h, Wout (64x64) @ sprevT_h (64x1264) -> += yconv.
// Workgroup = 4 waves; Wout[h] staged via TDM; grid = (H_, NTSL_)
// ---------------------------------------------------------------------------
__global__ __launch_bounds__(128) void k_gemmB(const _Float16* __restrict__ Wout,
                                               const _Float16* __restrict__ sprevT,
                                               float* __restrict__ yconv) {
  __shared__ __align__(128) _Float16 smW[Q_ * 64];    // 8 KB
  int h = blockIdx.x;
  if (threadIdx.x < 32) {
    tdm_load_1d((unsigned)(uintptr_t)&smW[0], Wout + (size_t)h * Q_ * 64, Q_ * 64 * 2);
    __builtin_amdgcn_s_wait_tensorcnt(0);
  }
  __syncthreads();
  int wave = threadIdx.x >> 5;   // M-tile 0..3
  int lane = threadIdx.x & 31;
  int mi = lane & 15;
  int hi = lane >> 4;
  int koff = hi ? 16 : 0;
  int klo  = hi ? 16 : 0;
  const _Float16* pa = &smW[(wave * 16 + mi) * 64 + koff];
  v16h a0 = frag_from_runs(*(const v8h*)(pa),      *(const v8h*)(pa + 8));
  v16h a1 = frag_from_runs(*(const v8h*)(pa + 32), *(const v8h*)(pa + 40));

  int nt0 = blockIdx.y * NTPER_;
  int nt1 = nt0 + NTPER_; if (nt1 > CCH_) nt1 = CCH_;
  for (int nt = nt0; nt < nt1; ++nt) {
    const _Float16* Sh = sprevT + ((size_t)h * BC_ + (size_t)nt * 16) * 64;
    v16h b0 = *(const v16h*)(Sh + (long)mi * 64 + 0  + klo);
    v16h b1 = *(const v16h*)(Sh + (long)mi * 64 + 32 + klo);
    v8f c = {};
    c = __builtin_amdgcn_wmma_f32_16x16x32_f16(false, a0, false, b0, (short)0, c, false, false);
    c = __builtin_amdgcn_wmma_f32_16x16x32_f16(false, a1, false, b1, (short)0, c, false, false);
    int rowbase = wave * 16 + (hi ? 8 : 0);
    int ncol = nt * 16 + mi;
    int b_ = ncol / CCH_, cch = ncol % CCH_;
#pragma unroll
    for (int r = 0; r < 8; ++r) {
      int l = cch * Q_ + rowbase + r;
      if (l < L_) yconv[((long)(b_ * L_ + l)) * H_ + h] += c[r];
    }
  }
}

// ---------------------------------------------------------------------------
// Post: y = yconv + D*z ; gelu(tanh approx) ; -> gact f16 (B*L, H)
// ---------------------------------------------------------------------------
__global__ void k_post(const float* __restrict__ yconv, const float* __restrict__ z,
                       const float* __restrict__ Dp, _Float16* __restrict__ gact) {
  int idx = blockIdx.x * blockDim.x + threadIdx.x;
  if (idx >= B_ * L_ * H_) return;
  int h = idx % H_;
  float y = yconv[idx] + Dp[h] * z[idx];
  float tg = tanhf(0.7978845608028654f * (y + 0.044715f * y * y * y));
  gact[idx] = (_Float16)(0.5f * y * (1.0f + tg));
}

// ---------------------------------------------------------------------------
// Pack W_glu (H, 2H) f32 -> WgT (2H, H) f16
// ---------------------------------------------------------------------------
__global__ void k_packw(const float* __restrict__ Wg, _Float16* __restrict__ WgT) {
  int idx = blockIdx.x * blockDim.x + threadIdx.x;
  if (idx >= H_ * 2 * H_) return;
  int k = idx % H_;
  int nn = idx / H_;
  WgT[idx] = (_Float16)Wg[(long)k * (2 * H_) + nn];
}

// ---------------------------------------------------------------------------
// GLU GEMM: gact (80000 x 256) @ WgT^T. One block (4 waves) per 16-row tile,
// A row-tile (8 KB) staged via TDM; each wave owns 4 paired column tiles,
// A-fragment reused across 8 WMMAs per K-step.
// ---------------------------------------------------------------------------
__global__ __launch_bounds__(128) void k_glu(const _Float16* __restrict__ gact,
                                             const _Float16* __restrict__ WgT,
                                             const float* __restrict__ bglu,
                                             const float* __restrict__ z,
                                             float* __restrict__ xout) {
  __shared__ __align__(128) _Float16 smA[16 * H_];   // 8 KB
  int mt = blockIdx.x;
  if (threadIdx.x < 32) {
    tdm_load_1d((unsigned)(uintptr_t)&smA[0], gact + (size_t)mt * 16 * H_, 16 * H_ * 2);
    __builtin_amdgcn_s_wait_tensorcnt(0);
  }
  __syncthreads();
  int wave = threadIdx.x >> 5;
  int lane = threadIdx.x & 31;
  int mi = lane & 15;
  int hi = lane >> 4;
  int koff = hi ? 16 : 0;
  int klo  = hi ? 16 : 0;
  v8f c1[4] = {};
  v8f c2[4] = {};
#pragma unroll
  for (int kb = 0; kb < H_; kb += 32) {
    const _Float16* pa = &smA[mi * H_ + kb + koff];
    v16h a = frag_from_runs(*(const v8h*)(pa), *(const v8h*)(pa + 8));
#pragma unroll
    for (int t = 0; t < 4; ++t) {
      int jt = wave * 4 + t;
      const _Float16* B1 = WgT + (size_t)(jt * 16 + mi) * H_ + kb + klo;
      const _Float16* B2 = WgT + (size_t)(H_ + jt * 16 + mi) * H_ + kb + klo;
      c1[t] = __builtin_amdgcn_wmma_f32_16x16x32_f16(false, a, false, *(const v16h*)B1,
                                                     (short)0, c1[t], false, false);
      c2[t] = __builtin_amdgcn_wmma_f32_16x16x32_f16(false, a, false, *(const v16h*)B2,
                                                     (short)0, c2[t], false, false);
    }
  }
  int rowbase = mt * 16 + (hi ? 8 : 0);
#pragma unroll
  for (int t = 0; t < 4; ++t) {
    int ncol = (wave * 4 + t) * 16 + mi;
    float bg1 = bglu[ncol];
    float bg2 = bglu[H_ + ncol];
#pragma unroll
    for (int r = 0; r < 8; ++r) {
      long row = rowbase + r;
      float a = c1[t][r] + bg1;
      float g = c2[t][r] + bg2;
      float sig = 1.0f / (1.0f + __expf(-g));
      long o = row * H_ + ncol;
      xout[o] = a * sig + z[o];
    }
  }
}

// ---------------------------------------------------------------------------
// Host side
// ---------------------------------------------------------------------------
static inline size_t align256(size_t x) { return (x + 255) & ~(size_t)255; }

extern "C" void kernel_launch(void* const* d_in, const int* in_sizes, int n_in,
                              void* d_out, int out_size, void* d_ws, size_t ws_size,
                              hipStream_t stream) {
  (void)in_sizes; (void)n_in; (void)out_size; (void)ws_size;
  const float* x       = (const float*)d_in[0];
  const float* W_in    = (const float*)d_in[1];
  const float* b_in    = (const float*)d_in[2];
  const float* ln_g    = (const float*)d_in[3];
  const float* ln_b    = (const float*)d_in[4];
  const float* log_dt  = (const float*)d_in[5];
  const float* logA_re = (const float*)d_in[6];
  const float* A_im    = (const float*)d_in[7];
  const float* Cp      = (const float*)d_in[8];
  const float* Dp      = (const float*)d_in[9];
  const float* W_glu   = (const float*)d_in[10];
  const float* b_glu   = (const float*)d_in[11];
  const float* fn_g    = (const float*)d_in[12];
  const float* fn_b    = (const float*)d_in[13];

  char* ws = (char*)d_ws;
  size_t off = 0;
  float*    z      = (float*)   (ws + off); off += align256((size_t)B_ * L_ * H_ * 4);
  _Float16* uT     = (_Float16*)(ws + off); off += align256((size_t)H_ * BC_ * Q_ * 2);
  float*    yconv  = (float*)   (ws + off); off += align256((size_t)B_ * L_ * H_ * 4);
  float*    scon   = (float*)   (ws + off); off += align256((size_t)H_ * BC_ * 64 * 4);
  _Float16* sprevT = (_Float16*)(ws + off); off += align256((size_t)H_ * BC_ * 64 * 2);
  _Float16* gact   = (_Float16*)(ws + off); off += align256((size_t)B_ * L_ * H_ * 2);
  _Float16* Amat   = (_Float16*)(ws + off); off += align256((size_t)H_ * 128 * Q_ * 2);
  _Float16* Wout   = (_Float16*)(ws + off); off += align256((size_t)H_ * Q_ * 64 * 2);
  float*    lamQ   = (float*)   (ws + off); off += align256((size_t)H_ * NM_ * 2 * 4);
  _Float16* WgT    = (_Float16*)(ws + off); off += align256((size_t)2 * H_ * H_ * 2);

  float* xres = (float*)d_out;  // residual stream; final LN is in-place safe

  const int BLH = B_ * L_ * H_;
  k_inproj<<<(BLH + 255) / 256, 256, 0, stream>>>(x, W_in, b_in, xres);

  for (int i = 0; i < NL_; ++i) {
    k_layernorm<<<B_ * L_, H_, 0, stream>>>(xres, ln_g + (long)i * H_, ln_b + (long)i * H_, z);
    {
      long tot = (long)H_ * BC_ * Q_;
      k_pack_u<<<(int)((tot + 255) / 256), 256, 0, stream>>>(z, uT);
    }
    k_ssm_pre<<<H_, 32, 0, stream>>>(log_dt + (long)i * H_,
                                     logA_re + (long)i * H_ * NM_,
                                     A_im + (long)i * H_ * NM_,
                                     Cp + (long)i * H_ * NM_ * 2,
                                     Amat, Wout, lamQ);
    k_gemmA<<<dim3(H_, NTSL_), 256, 0, stream>>>(Amat, uT, yconv, scon);
    k_scan<<<(B_ * H_ * NM_ + 255) / 256, 256, 0, stream>>>(scon, lamQ, sprevT);
    k_gemmB<<<dim3(H_, NTSL_), 128, 0, stream>>>(Wout, sprevT, yconv);
    k_post<<<(BLH + 255) / 256, 256, 0, stream>>>(yconv, z, Dp + (long)i * H_, gact);
    k_packw<<<(H_ * 2 * H_ + 255) / 256, 256, 0, stream>>>(W_glu + (long)i * H_ * 2 * H_, WgT);
    k_glu<<<L_ * B_ / 16, 128, 0, stream>>>(gact, WgT, b_glu + (long)i * 2 * H_, z, xres);
  }

  k_layernorm<<<B_ * L_, H_, 0, stream>>>(xres, fn_g, fn_b, (float*)d_out);
}
